// ConvTransBlock_28724741276384
// MI455X (gfx1250) — matrix operand order repaired and checked by reading, not compiled, not run on hardware
//
#include <hip/hip_runtime.h>
#include <hip/hip_bf16.h>
#include <math.h>

typedef __bf16 bf16_t;
typedef __attribute__((ext_vector_type(16))) __bf16 v16bf;
typedef __attribute__((ext_vector_type(8)))  __bf16 v8bf;
typedef __attribute__((ext_vector_type(4)))  __bf16 v4bf;
typedef __attribute__((ext_vector_type(8)))  float  v8f;
typedef __attribute__((ext_vector_type(4)))  float  v4f;
typedef __attribute__((ext_vector_type(4)))  unsigned tdm_u4;
typedef __attribute__((ext_vector_type(8)))  int      tdm_i8;
typedef __attribute__((ext_vector_type(4)))  int      tdm_i4;

#define M_ROWS   131072      // B*H*W
#define SCALEF   0.17677669529663687f

// ---------- helpers ----------
__device__ __forceinline__ bf16_t f2bf(float f) {
    unsigned u = __float_as_uint(f);
    unsigned r = u + 0x7FFFu + ((u >> 16) & 1u);
    unsigned short h = (unsigned short)(r >> 16);
    return __builtin_bit_cast(bf16_t, h);
}

// A-operand (16x32, 16-bit): lane m=l&15; K chunks at khi*8 and 16+khi*8
__device__ __forceinline__ v16bf load_mk(const bf16_t* base, int ld, int lane) {
    const bf16_t* p = base + (size_t)(lane & 15) * ld + ((lane >> 4) << 3);
    union { v16bf v; v8bf h[2]; } u;
    u.h[0] = *(const v8bf*)(p);
    u.h[1] = *(const v8bf*)(p + 16);
    return u.v;
}
__device__ __forceinline__ v16bf load_mk_guard(const bf16_t* base, int ld, int lane, bool valid) {
    union { v16bf v; v8bf h[2]; unsigned w[8]; } u;
    if (valid) {
        const bf16_t* p = base + (size_t)(lane & 15) * ld + ((lane >> 4) << 3);
        u.h[0] = *(const v8bf*)(p);
        u.h[1] = *(const v8bf*)(p + 16);
    } else {
#pragma unroll
        for (int i = 0; i < 8; ++i) u.w[i] = 0;
    }
    return u.v;
}
// B-operand (32x16, 16-bit) from row-major [n][k] source: lane n=l&15; K chunks at khi*8 and 16+khi*8
__device__ __forceinline__ v16bf load_nk(const bf16_t* base, int ld, int lane) {
    const bf16_t* p = base + (size_t)(lane & 15) * ld + ((lane >> 4) << 3);
    union { v16bf v; v8bf h[2]; } u;
    u.h[0] = *(const v8bf*)(p);
    u.h[1] = *(const v8bf*)(p + 16);
    return u.v;
}
__device__ __forceinline__ v8f wmma_bf16(v16bf a, v16bf b, v8f c) {
    return __builtin_amdgcn_wmma_f32_16x16x32_bf16(false, a, false, b, (short)0, c, false, false);
}

// ---------- Tensor Data Mover: issue a 2D tile load (global -> LDS) ----------
// D# per CDNA5 ISA: group0 {count=1, lds_addr, global_addr, type=2},
// group1 {data_size=2B, tensor_dims == tile dims (interior tile), dim0 stride}.
// amdgpu-toolchain (clang-23) 6-arg builtin: (u32x4, i32x8, i32x4, i32x4, i32x8, i32 cpol).
__device__ __forceinline__ void tdm_load_2d(unsigned lds_off, const void* gaddr,
                                            unsigned d0, unsigned d1,
                                            unsigned long long stride0) {
    unsigned long long ga = (unsigned long long)(uintptr_t)gaddr;
    tdm_u4 g0;
    g0[0] = 1u;                                            // count=1 (valid user D#)
    g0[1] = lds_off;                                       // LDS byte address
    g0[2] = (unsigned)(ga & 0xffffffffu);                  // global_addr[31:0]
    g0[3] = (unsigned)((ga >> 32) & 0x01ffffffu) | (2u << 30); // addr[56:32] | type=2
    tdm_i8 g1;
    g1[0] = (int)(1u << 16);                               // data_size code 1 = 2 bytes
    g1[1] = (int)((d0 & 0xffffu) << 16);                   // tensor_dim0[15:0]
    g1[2] = (int)((d1 & 0xffffu) << 16);                   // dim0[31:16]=0 | tensor_dim1[15:0]
    g1[3] = (int)((d0 & 0xffffu) << 16);                   // dim1[31:16]=0 | tile_dim0
    g1[4] = (int)(d1 & 0xffffu);                           // tile_dim1 | tile_dim2=0
    g1[5] = (int)(stride0 & 0xffffffffu);                  // tensor_dim0_stride[31:0]
    g1[6] = (int)((stride0 >> 32) & 0xffffu);              // stride[47:32] | dim1_stride lo=0
    g1[7] = 0;
    tdm_i4 gz4 = {0, 0, 0, 0};
    tdm_i8 gz8 = {0, 0, 0, 0, 0, 0, 0, 0};
    __builtin_amdgcn_tensor_load_to_lds(g0, g1, gz4, gz4, gz8, 0);
}

// ---------- elementwise fp32 -> bf16 ----------
__global__ __launch_bounds__(256) void cvt_bf16(const float* __restrict__ in,
                                                bf16_t* __restrict__ out, int n) {
    int i = (blockIdx.x * 256 + threadIdx.x) * 4;
    if (i >= n) return;
    v4f x = *(const v4f*)(in + i);
    v4bf o;
#pragma unroll
    for (int c = 0; c < 4; ++c) o[c] = f2bf(x[c]);
    *(v4bf*)(out + i) = o;
}

// ---------- pack weight [K][N] (row-major) into WMMA-B tile order ----------
// packed[((kt*(N/16)+nt)*32 + lane)*16 + e] = W[(kt*32 + (e<8 ? (lane>>4)*8 + e : 16 + (lane>>4)*8 + e-8))*N + nt*16 + (lane&15)]
__global__ __launch_bounds__(256) void pack_w(const float* __restrict__ W,
                                              bf16_t* __restrict__ out, int K, int N) {
    int idx = blockIdx.x * 256 + threadIdx.x;
    if (idx >= K * N) return;
    int e    = idx & 15;
    int lane = (idx >> 4) & 31;
    int rest = idx >> 9;
    int NT   = N >> 4;
    int nt   = rest % NT;
    int kt   = rest / NT;
    int k = kt * 32 + (e < 8 ? ((lane >> 4) << 3) + e : 16 + ((lane >> 4) << 3) + (e - 8));
    int n = nt * 16 + (lane & 15);
    out[idx] = f2bf(W[(size_t)k * N + n]);
}

// ---------- LayerNorm (one wave per 128-channel row) -> bf16 ----------
__global__ __launch_bounds__(256) void ln_kernel(const float* __restrict__ in,
                                                 const float* __restrict__ g,
                                                 const float* __restrict__ b,
                                                 bf16_t* __restrict__ out) {
    int lane = threadIdx.x & 31;
    size_t row = (size_t)blockIdx.x * 8 + (threadIdx.x >> 5);
    const float* p = in + row * 128 + lane * 4;
    v4f x = *(const v4f*)p;
    float s  = x[0] + x[1] + x[2] + x[3];
    float s2 = x[0]*x[0] + x[1]*x[1] + x[2]*x[2] + x[3]*x[3];
#pragma unroll
    for (int off = 1; off < 32; off <<= 1) {
        s  += __shfl_xor(s,  off, 32);
        s2 += __shfl_xor(s2, off, 32);
    }
    float mu   = s * (1.f / 128.f);
    float var  = s2 * (1.f / 128.f) - mu * mu;
    float rstd = rsqrtf(var + 1e-5f);
    v4bf o;
#pragma unroll
    for (int c = 0; c < 4; ++c)
        o[c] = f2bf((x[c] - mu) * rstd * g[lane * 4 + c] + b[lane * 4 + c]);
    *(v4bf*)(out + row * 128 + lane * 4) = o;
}

// ---------- generic bf16 WMMA GEMM with TDM-staged, double-buffered LDS operands ----------
// 8 waves, wave tile 32x64, WG tile 128x128. Wave 0 drives the Tensor Data Mover.
enum { EPI_SPLIT = 0, EPI_QKV = 1, EPI_PROJ = 2, EPI_GELU = 3, EPI_MLP2 = 4, EPI_FINAL = 5 };

template <int EPI>
__global__ __launch_bounds__(256) void gemm_bf16(
    const bf16_t* __restrict__ A, const bf16_t* __restrict__ Bp,
    const float* __restrict__ bias, int K, int N, int ldA,
    const float* __restrict__ res, float* __restrict__ outF,
    float* __restrict__ outF2, bf16_t* __restrict__ outB) {
    __shared__ __align__(16) bf16_t sA[2][128 * 32];   // A panel: 128 rows x 32 K
    __shared__ __align__(16) bf16_t sB[2][8 * 512];    // B panel: 8 n-tiles x 512 (packed)
    const int lane = threadIdx.x & 31;
    const int wv   = threadIdx.x >> 5;
    const int mblk = blockIdx.x * 128;
    const int m0l  = (wv & 3) * 32;                    // row offset inside LDS panel
    const int n0l  = (wv >> 2) * 64;                   // col offset inside 128-wide slab
    const int NT16 = N >> 4;
    const int ntB  = (blockIdx.y * 128) >> 4;          // first packed n-tile of this slab
    const int nKt  = K >> 5;

    v8f zero = {0.f, 0.f, 0.f, 0.f, 0.f, 0.f, 0.f, 0.f};
    v8f acc[2][4];
#pragma unroll
    for (int i = 0; i < 2; ++i)
#pragma unroll
        for (int j = 0; j < 4; ++j) acc[i][j] = zero;

    // prologue: TDM-load k-step 0
    if (wv == 0) {
        tdm_load_2d((unsigned)(uintptr_t)&sA[0][0], A + (size_t)mblk * ldA,
                    32u, 128u, (unsigned long long)ldA);
        tdm_load_2d((unsigned)(uintptr_t)&sB[0][0], Bp + ((size_t)ntB) * 512,
                    4096u, 1u, 4096ull);
    }
    for (int kt = 0; kt < nKt; ++kt) {
        int buf = kt & 1;
        if (wv == 0) {
            if (kt + 1 < nKt) {   // prefetch next k-step into the other buffer
                tdm_load_2d((unsigned)(uintptr_t)&sA[buf ^ 1][0],
                            A + (size_t)mblk * ldA + (kt + 1) * 32,
                            32u, 128u, (unsigned long long)ldA);
                tdm_load_2d((unsigned)(uintptr_t)&sB[buf ^ 1][0],
                            Bp + ((size_t)(kt + 1) * NT16 + ntB) * 512,
                            4096u, 1u, 4096ull);
                __builtin_amdgcn_s_wait_tensorcnt(2);  // pair(kt) done, pair(kt+1) in flight
            } else {
                __builtin_amdgcn_s_wait_tensorcnt(0);
            }
        }
        __syncthreads();                               // publish buf to all waves
        v16bf a0 = load_mk(sA[buf] + (size_t)(m0l) * 32, 32, lane);
        v16bf a1 = load_mk(sA[buf] + (size_t)(m0l + 16) * 32, 32, lane);
#pragma unroll
        for (int nt = 0; nt < 4; ++nt) {
            const bf16_t* bp = sB[buf] + (((size_t)(n0l >> 4) + nt) * 32 + lane) * 16;
            union { v16bf v; v8bf h[2]; } ub;
            ub.h[0] = *(const v8bf*)(bp);
            ub.h[1] = *(const v8bf*)(bp + 8);
            acc[0][nt] = wmma_bf16(a0, ub.v, acc[0][nt]);
            acc[1][nt] = wmma_bf16(a1, ub.v, acc[1][nt]);
        }
        __syncthreads();                               // all reads done before buf reuse
    }

    const int hi = lane >> 4, lo = lane & 15;
#pragma unroll
    for (int mt = 0; mt < 2; ++mt)
#pragma unroll
        for (int nt = 0; nt < 4; ++nt)
#pragma unroll
            for (int r = 0; r < 8; ++r) {
                size_t row = (size_t)(mblk + m0l + mt * 16 + r + hi * 8);
                int col = blockIdx.y * 128 + n0l + nt * 16 + lo;
                float v = acc[mt][nt][r] + bias[col];
                if (EPI == EPI_SPLIT) {
                    if (col < 128) {
                        outF[row * 128 + col] = v;          // conv_x fp32
                        outB[row * 128 + col] = f2bf(v);    // conv_x bf16
                    } else {
                        outF2[row * 128 + (col - 128)] = v; // trans_x fp32
                    }
                } else if (EPI == EPI_QKV) {
                    outB[row * 384 + col] = f2bf(v);
                } else if (EPI == EPI_PROJ) {
                    outF[row * 128 + col] = v + res[row * 128 + col]; // t = trans_x + proj
                } else if (EPI == EPI_GELU) {
                    float gel = 0.5f * v * (1.f + erff(v * 0.70710678118654752f));
                    outB[row * 512 + col] = f2bf(gel);
                } else if (EPI == EPI_MLP2) {
                    outB[row * 256 + 128 + col] = f2bf(v + res[row * 128 + col]); // cat[:,128:]
                } else { // EPI_FINAL
                    outF[row * 256 + col] = v + res[row * 256 + col]; // x + res
                }
            }
}

// ---------- 3x3 conv as implicit GEMM (K = 9 taps x 128), SAME zero padding ----------
template <int SECOND>
__global__ __launch_bounds__(256) void conv3_bf16(
    const bf16_t* __restrict__ A, const bf16_t* __restrict__ Bp,
    const float* __restrict__ bias, const float* __restrict__ convx,
    bf16_t* __restrict__ outB) {
    const int lane = threadIdx.x & 31;
    const int wv   = threadIdx.x >> 5;
    const int m0   = blockIdx.x * 128 + (wv & 3) * 32;
    const int n0   = (wv >> 2) * 64;
    v8f zero = {0.f, 0.f, 0.f, 0.f, 0.f, 0.f, 0.f, 0.f};
    v8f acc[2][4];
#pragma unroll
    for (int i = 0; i < 2; ++i)
#pragma unroll
        for (int j = 0; j < 4; ++j) acc[i][j] = zero;

    for (int tap = 0; tap < 9; ++tap) {
        int dy = tap / 3 - 1, dx = tap % 3 - 1;
        int dlt = dy * 256 + dx;
        for (int kl = 0; kl < 4; ++kl) {       // 128 channels per tap
            v16bf a[2];
#pragma unroll
            for (int mt = 0; mt < 2; ++mt) {
                int m = m0 + mt * 16 + (lane & 15);
                int x = (m & 255) + dx;
                int y = ((m >> 8) & 255) + dy;
                bool valid = ((unsigned)x < 256u) && ((unsigned)y < 256u);
                a[mt] = load_mk_guard(A + (size_t)(m0 + mt * 16 + dlt) * 128 + kl * 32,
                                      128, lane, valid);
            }
            int ktB = tap * 4 + kl;
#pragma unroll
            for (int nt = 0; nt < 4; ++nt) {
                const bf16_t* bp = Bp + (((size_t)ktB * 8 + (n0 >> 4) + nt) * 32 + lane) * 16;
                union { v16bf v; v8bf h[2]; } ub;
                ub.h[0] = *(const v8bf*)(bp);
                ub.h[1] = *(const v8bf*)(bp + 8);
                acc[0][nt] = wmma_bf16(a[0], ub.v, acc[0][nt]);
                acc[1][nt] = wmma_bf16(a[1], ub.v, acc[1][nt]);
            }
        }
    }
    const int hi = lane >> 4, lo = lane & 15;
#pragma unroll
    for (int mt = 0; mt < 2; ++mt)
#pragma unroll
        for (int nt = 0; nt < 4; ++nt)
#pragma unroll
            for (int r = 0; r < 8; ++r) {
                size_t row = (size_t)(m0 + mt * 16 + r + hi * 8);
                int col = n0 + nt * 16 + lo;
                float v = acc[mt][nt][r] + bias[col];
                v = (v >= 0.f) ? v : 0.01f * v;                 // lrelu
                if (SECOND) {
                    v += 2.f * convx[row * 128 + col];          // r + 2*conv_x
                    outB[row * 256 + col] = f2bf(v);            // cat[:, :128]
                } else {
                    outB[row * 128 + col] = f2bf(v);            // r1
                }
            }
}

// ---------- shifted-window attention: one wave per (batch, window, head) ----------
__global__ __launch_bounds__(32) void attn_kernel(const bf16_t* __restrict__ qkv,
                                                  const float* __restrict__ rpb,
                                                  bf16_t* __restrict__ ao) {
    __shared__ __align__(16) bf16_t ks[64 * 32];  // K rows [token][d]
    __shared__ __align__(16) bf16_t vT[32 * 64];  // V transposed [d][token]
    __shared__ __align__(16) bf16_t Ps[16 * 64];  // softmax block, A-layout staging
    const int lane = threadIdx.x;
    int gid = blockIdx.x;
    int h  = gid & 3;
    int w  = (gid >> 2) & 1023;
    int bb = gid >> 12;
    int wh = w >> 5, wwn = w & 31;

    for (int t = lane; t < 64; t += 32) {
        int sy = (wh * 8 + (t >> 3) + 4) & 255;
        int sx = (wwn * 8 + (t & 7) + 4) & 255;
        const bf16_t* src = qkv + (size_t)(bb * 65536 + sy * 256 + sx) * 384 + 128 + h * 32;
#pragma unroll
        for (int c = 0; c < 4; ++c)
            *(v8bf*)(ks + t * 32 + c * 8) = *(const v8bf*)(src + c * 8);
        const bf16_t* srcv = src + 128;
#pragma unroll
        for (int c = 0; c < 4; ++c) {
            v8bf vv = *(const v8bf*)(srcv + c * 8);
#pragma unroll
            for (int e = 0; e < 8; ++e) vT[(c * 8 + e) * 64 + t] = vv[e];
        }
    }
    __syncthreads();

    const int hi = lane >> 4, lo = lane & 15;
    for (int ti = 0; ti < 4; ++ti) {
        // Q tile directly from global in A-layout (HD=32 -> single K step)
        int qi = ti * 16 + lo;
        int qy = (wh * 8 + (qi >> 3) + 4) & 255;
        int qx = (wwn * 8 + (qi & 7) + 4) & 255;
        const bf16_t* qp = qkv + (size_t)(bb * 65536 + qy * 256 + qx) * 384 + h * 32 + hi * 8;
        union { v16bf v; v8bf c2[2]; } qa;
        qa.c2[0] = *(const v8bf*)(qp);
        qa.c2[1] = *(const v8bf*)(qp + 16);

        v8f zero = {0.f, 0.f, 0.f, 0.f, 0.f, 0.f, 0.f, 0.f};
        v8f S[4];
#pragma unroll
        for (int tj = 0; tj < 4; ++tj) {
            v16bf bk = load_nk(ks + (size_t)(tj * 16) * 32, 32, lane);
            S[tj] = wmma_bf16(qa.v, bk, zero);
        }
        // bias + shift-mask + softmax (rows live across 16-lane halves)
#pragma unroll
        for (int r = 0; r < 8; ++r) {
            int i  = ti * 16 + r + hi * 8;
            int iy = i >> 3, ix = i & 7;
            int ayy = wh * 8 + iy, axx = wwn * 8 + ix;
            int idi = ((ayy >= 248) + (ayy >= 252)) * 3 + ((axx >= 248) + (axx >= 252));
            float vals[4];
            float mxr = -3e38f;
#pragma unroll
            for (int tj = 0; tj < 4; ++tj) {
                int j  = tj * 16 + lo;
                int jy = j >> 3, jx = j & 7;
                int byy = wh * 8 + jy, bxx = wwn * 8 + jx;
                int idj = ((byy >= 248) + (byy >= 252)) * 3 + ((bxx >= 248) + (bxx >= 252));
                int rel = (iy - jy + 7) * 15 + (ix - jx + 7);
                float sv = S[tj][r] * SCALEF + rpb[rel * 4 + h];
                if (idi != idj) sv -= 1e9f;
                vals[tj] = sv;
                mxr = fmaxf(mxr, sv);
            }
#pragma unroll
            for (int off = 1; off < 16; off <<= 1)
                mxr = fmaxf(mxr, __shfl_xor(mxr, off, 32));
            float sum = 0.f;
#pragma unroll
            for (int tj = 0; tj < 4; ++tj) {
                vals[tj] = __expf(vals[tj] - mxr);
                sum += vals[tj];
            }
#pragma unroll
            for (int off = 1; off < 16; off <<= 1)
                sum += __shfl_xor(sum, off, 32);
            float inv = 1.f / sum;
#pragma unroll
            for (int tj = 0; tj < 4; ++tj)
                Ps[(r + hi * 8) * 64 + tj * 16 + lo] = f2bf(vals[tj] * inv);
        }
        __syncthreads();
        // P @ V : out 16x32
        v8f O[2] = {zero, zero};
#pragma unroll
        for (int kt2 = 0; kt2 < 2; ++kt2) {
            v16bf pa = load_mk(Ps + kt2 * 32, 64, lane);
#pragma unroll
            for (int nt = 0; nt < 2; ++nt) {
                v16bf bv = load_nk(vT + (size_t)(nt * 16) * 64 + kt2 * 32, 64, lane);
                O[nt] = wmma_bf16(pa, bv, O[nt]);
            }
        }
#pragma unroll
        for (int r = 0; r < 8; ++r) {
            int i  = ti * 16 + r + hi * 8;
            int sy = (wh * 8 + (i >> 3) + 4) & 255;
            int sx = (wwn * 8 + (i & 7) + 4) & 255;
            size_t px = (size_t)(bb * 65536 + sy * 256 + sx);
#pragma unroll
            for (int nt = 0; nt < 2; ++nt)
                ao[px * 128 + h * 32 + nt * 16 + lo] = f2bf(O[nt][r]);
        }
        __syncthreads();
    }
}

// ---------- host orchestration ----------
extern "C" void kernel_launch(void* const* d_in, const int* in_sizes, int n_in,
                              void* d_out, int out_size, void* d_ws, size_t ws_size,
                              hipStream_t stream) {
    const float* x    = (const float*)d_in[0];
    const float* c1w  = (const float*)d_in[1];
    const float* c1b  = (const float*)d_in[2];
    const float* rw1  = (const float*)d_in[3];
    const float* rb1  = (const float*)d_in[4];
    const float* rw2  = (const float*)d_in[5];
    const float* rb2  = (const float*)d_in[6];
    const float* ln1g = (const float*)d_in[7];
    const float* ln1b = (const float*)d_in[8];
    const float* qkvw = (const float*)d_in[9];
    const float* qkvb = (const float*)d_in[10];
    const float* rpb  = (const float*)d_in[11];
    const float* pw   = (const float*)d_in[12];
    const float* pb   = (const float*)d_in[13];
    const float* ln2g = (const float*)d_in[14];
    const float* ln2b = (const float*)d_in[15];
    const float* mw1  = (const float*)d_in[16];
    const float* mb1  = (const float*)d_in[17];
    const float* mw2  = (const float*)d_in[18];
    const float* mb2  = (const float*)d_in[19];
    const float* c2w  = (const float*)d_in[20];
    const float* c2b  = (const float*)d_in[21];
    float* out = (float*)d_out;
    char*  ws  = (char*)d_ws;
    const size_t M = M_ROWS;

    // workspace layout (bytes)
    size_t o = 0;
    bf16_t* xb      = (bf16_t*)(ws + o); o += M * 256 * 2;
    float*  convx_f = (float*)(ws + o);  o += M * 128 * 4;
    float*  transx_f= (float*)(ws + o);  o += M * 128 * 4;
    bf16_t* convx_b = (bf16_t*)(ws + o); o += M * 128 * 2;
    bf16_t* r1_b    = (bf16_t*)(ws + o); o += M * 128 * 2;
    bf16_t* cat_b   = (bf16_t*)(ws + o); o += M * 256 * 2;
    bf16_t* ln_b    = (bf16_t*)(ws + o); o += M * 128 * 2;
    bf16_t* qkv_b   = (bf16_t*)(ws + o); o += M * 384 * 2;
    bf16_t* ao_b    = (bf16_t*)(ws + o); o += M * 128 * 2;
    float*  t_f     = (float*)(ws + o);  o += M * 128 * 4;
    bf16_t* g_b     = (bf16_t*)(ws + o); o += M * 512 * 2;
    bf16_t* c1w_p   = (bf16_t*)(ws + o); o += 65536 * 2;
    bf16_t* rw1_p   = (bf16_t*)(ws + o); o += 147456 * 2;
    bf16_t* rw2_p   = (bf16_t*)(ws + o); o += 147456 * 2;
    bf16_t* qkvw_p  = (bf16_t*)(ws + o); o += 49152 * 2;
    bf16_t* pw_p    = (bf16_t*)(ws + o); o += 16384 * 2;
    bf16_t* mw1_p   = (bf16_t*)(ws + o); o += 65536 * 2;
    bf16_t* mw2_p   = (bf16_t*)(ws + o); o += 65536 * 2;
    bf16_t* c2w_p   = (bf16_t*)(ws + o); o += 65536 * 2;

    // stage 0: convert input + pack weights
    cvt_bf16<<<(M * 256 / 4 + 255) / 256, 256, 0, stream>>>(x, xb, (int)(M * 256));
    pack_w<<<256, 256, 0, stream>>>(c1w,  c1w_p,  256, 256);
    pack_w<<<576, 256, 0, stream>>>(rw1,  rw1_p, 1152, 128);
    pack_w<<<576, 256, 0, stream>>>(rw2,  rw2_p, 1152, 128);
    pack_w<<<192, 256, 0, stream>>>(qkvw, qkvw_p, 128, 384);
    pack_w<<<64,  256, 0, stream>>>(pw,   pw_p,   128, 128);
    pack_w<<<256, 256, 0, stream>>>(mw1,  mw1_p,  128, 512);
    pack_w<<<256, 256, 0, stream>>>(mw2,  mw2_p,  512, 128);
    pack_w<<<256, 256, 0, stream>>>(c2w,  c2w_p,  256, 256);

    // stage 1: conv1 1x1 + channel split
    gemm_bf16<EPI_SPLIT><<<dim3(1024, 2), 256, 0, stream>>>(
        xb, c1w_p, c1b, 256, 256, 256, nullptr, convx_f, transx_f, convx_b);

    // stage 2: conv branch (two 3x3 convs, lrelu; second adds 2*conv_x -> cat[:, :128])
    conv3_bf16<0><<<1024, 256, 0, stream>>>(convx_b, rw1_p, rb1, nullptr, r1_b);
    conv3_bf16<1><<<1024, 256, 0, stream>>>(r1_b,    rw2_p, rb2, convx_f, cat_b);

    // stage 3: transformer branch
    ln_kernel<<<16384, 256, 0, stream>>>(transx_f, ln1g, ln1b, ln_b);
    gemm_bf16<EPI_QKV><<<dim3(1024, 3), 256, 0, stream>>>(
        ln_b, qkvw_p, qkvb, 128, 384, 128, nullptr, nullptr, nullptr, qkv_b);
    attn_kernel<<<8192, 32, 0, stream>>>(qkv_b, rpb, ao_b);
    gemm_bf16<EPI_PROJ><<<dim3(1024, 1), 256, 0, stream>>>(
        ao_b, pw_p, pb, 128, 128, 128, transx_f, t_f, nullptr, nullptr);
    ln_kernel<<<16384, 256, 0, stream>>>(t_f, ln2g, ln2b, ln_b);
    gemm_bf16<EPI_GELU><<<dim3(1024, 4), 256, 0, stream>>>(
        ln_b, mw1_p, mb1, 128, 512, 128, nullptr, nullptr, nullptr, g_b);
    gemm_bf16<EPI_MLP2><<<dim3(1024, 1), 256, 0, stream>>>(
        g_b, mw2_p, mb2, 512, 128, 512, t_f, nullptr, nullptr, cat_b);

    // stage 4: final 1x1 conv on concat + residual with original x
    gemm_bf16<EPI_FINAL><<<dim3(1024, 2), 256, 0, stream>>>(
        cat_b, c2w_p, c2b, 256, 256, 256, x, out, nullptr, nullptr);

    (void)in_sizes; (void)n_in; (void)out_size; (void)ws_size;
}
